// Mamba_33543694582347
// MI455X (gfx1250) — matrix operand order repaired and checked by reading, not compiled, not run on hardware
//
#include <hip/hip_runtime.h>
#include <hip/hip_bf16.h>

typedef __attribute__((ext_vector_type(16))) __bf16 v16bf;
typedef __attribute__((ext_vector_type(8)))  __bf16 v8bf;
typedef __attribute__((ext_vector_type(8)))  float  v8f;
typedef __attribute__((ext_vector_type(4)))  int    v4i;

#define BM 128
#define BN 128
#define BK 32
#define LDSS 40   // padded LDS row stride (elements): 80B rows keep 16B alignment

// ---- CDNA5 async global->LDS path (guarded; falls back to load+store) ----
#if __has_builtin(__builtin_amdgcn_global_load_async_to_lds_b128)
#define USE_ASYNC 1
#else
#define USE_ASYNC 0
#endif
#if __has_builtin(__builtin_amdgcn_s_wait_asynccnt)
#define HAVE_WAIT_ASYNC_BUILTIN 1
#else
#define HAVE_WAIT_ASYNC_BUILTIN 0
#endif

typedef __attribute__((address_space(1))) v4i* g_v4i;  // global (prints as __device__)
typedef __attribute__((address_space(3))) v4i* l_v4i;  // LDS    (prints as __shared__)

__device__ __forceinline__ void cp16(void* lds_dst, const void* gsrc) {
#if USE_ASYNC
    // generic->as1 is numerically identity; generic LDS ptr low 32 bits == LDS offset
    __builtin_amdgcn_global_load_async_to_lds_b128(
        (g_v4i)(unsigned long long)gsrc,
        (l_v4i)(unsigned int)(unsigned long long)lds_dst,
        0, 0);
#else
    *(v8bf*)lds_dst = *(const v8bf*)gsrc;
#endif
}

__device__ __forceinline__ void wait_stage() {
#if USE_ASYNC
#if HAVE_WAIT_ASYNC_BUILTIN
    __builtin_amdgcn_s_wait_asynccnt(0);
#else
    asm volatile("s_wait_asynccnt 0x0" ::: "memory");
#endif
#endif
}

// ---------------------------------------------------------------------------
// Prep kernels
// ---------------------------------------------------------------------------
__global__ __launch_bounds__(256) void f32_to_bf16_k(const float* __restrict__ in,
                                                     __bf16* __restrict__ out, int n) {
    int i = blockIdx.x * 256 + threadIdx.x;
    if (i < n) out[i] = (__bf16)in[i];
}

// W (K x N row-major, f32) -> Wt (N x K row-major, bf16)
__global__ __launch_bounds__(256) void transpose_bf16_k(const float* __restrict__ W,
                                                        __bf16* __restrict__ Wt,
                                                        int K, int N) {
    int i = blockIdx.x * 256 + threadIdx.x;
    if (i < K * N) {
        int k = i / N, n = i % N;
        Wt[(size_t)n * K + k] = (__bf16)W[i];
    }
}

// ---------------------------------------------------------------------------
// GEMM: C(MxN, f32) = A(MxK, bf16 row-major) * Bt(NxK, bf16 row-major)^T
// 256 threads = 8 wave32; block tile 128x128; wave tile 32x64 (2x4 WMMA);
// double-buffered LDS fed by async global->LDS copies.
// ---------------------------------------------------------------------------
__global__ __launch_bounds__(256)
void gemm_bf16_wmma(const __bf16* __restrict__ A, const __bf16* __restrict__ Bt,
                    float* __restrict__ C, int M, int N, int K) {
    __shared__ __attribute__((aligned(16))) __bf16 As[2][BM][LDSS];
    __shared__ __attribute__((aligned(16))) __bf16 Bs[2][BN][LDSS];

    const int tid  = threadIdx.x;
    const int wave = tid >> 5;
    const int lane = tid & 31;
    const int h    = lane >> 4;   // half-wave select (ISA 7.12.2 16-bit layouts)
    const int lr   = lane & 15;

    const int m0 = blockIdx.y * BM;
    const int n0 = blockIdx.x * BN;
    const int wm = (wave >> 1) * 32;   // 0,32,64,96
    const int wn = (wave & 1) * 64;    // 0,64

    v8f acc[2][4] = {};

    // stage one K-tile into LDS buffer `buf`
    auto stage = [&](int kb, int buf) {
#pragma unroll
        for (int i = 0; i < 2; ++i) {           // A tile: 512 x 16B chunks
            int c = tid + i * 256;
            int row = c >> 2, kc = (c & 3) * 8;
            cp16(&As[buf][row][kc], A + (size_t)(m0 + row) * K + kb + kc);
        }
#pragma unroll
        for (int i = 0; i < 2; ++i) {           // B tile: 512 x 16B chunks
            int c = tid + i * 256;
            int row = c >> 2, kc = (c & 3) * 8;
            cp16(&Bs[buf][row][kc], Bt + (size_t)(n0 + row) * K + kb + kc);
        }
    };

    stage(0, 0);
    int cur = 0;

    for (int kb = 0; kb < K; kb += BK) {
        wait_stage();
        __syncthreads();                 // buf[cur] full; prior readers of buf[cur^1] done
        if (kb + BK < K) stage(kb + BK, cur ^ 1);

        union { v16bf v; v8bf p[2]; } afrag[2], bfrag[4];
#pragma unroll
        for (int mi = 0; mi < 2; ++mi) {
            int row = wm + mi * 16 + lr;                       // A row (M)
            afrag[mi].p[0] = *(const v8bf*)&As[cur][row][h * 8];      // K = h*8+0..7
            afrag[mi].p[1] = *(const v8bf*)&As[cur][row][16 + h * 8]; // K = 16+h*8+0..7
        }
#pragma unroll
        for (int ni = 0; ni < 4; ++ni) {
            int row = wn + ni * 16 + lr;                       // B col (N)
            bfrag[ni].p[0] = *(const v8bf*)&Bs[cur][row][h * 8];      // K = h*8+0..7
            bfrag[ni].p[1] = *(const v8bf*)&Bs[cur][row][16 + h * 8]; // K = 16+h*8+0..7
        }
#pragma unroll
        for (int mi = 0; mi < 2; ++mi)
#pragma unroll
            for (int ni = 0; ni < 4; ++ni)
                acc[mi][ni] = __builtin_amdgcn_wmma_f32_16x16x32_bf16(
                    false, afrag[mi].v, false, bfrag[ni].v,
                    (short)0, acc[mi][ni], false, false);
        cur ^= 1;
    }

    // C/D layout: lane -> col (lane&15), VGPR r -> row r + 8*h
#pragma unroll
    for (int mi = 0; mi < 2; ++mi)
#pragma unroll
        for (int ni = 0; ni < 4; ++ni) {
            int col = n0 + wn + ni * 16 + lr;
#pragma unroll
            for (int r = 0; r < 8; ++r) {
                int rowm = m0 + wm + mi * 16 + r + 8 * h;
                C[(size_t)rowm * N + col] = acc[mi][ni][r];
            }
        }
}

// ---------------------------------------------------------------------------
// Fused causal depthwise conv + selective scan + SiLU gating
// ---------------------------------------------------------------------------
#define LQ 2048
#define DI 2048
#define DS 16

__global__ __launch_bounds__(256)
void mamba_scan_k(const float* __restrict__ xz, const float* __restrict__ conv_w,
                  const float* __restrict__ conv_b, const float* __restrict__ A_log,
                  const float* __restrict__ Dp, __bf16* __restrict__ yb) {
    const int d = blockIdx.x * 256 + threadIdx.x;   // 0..2047
    const int b = blockIdx.y;

    float Ad[DS];
#pragma unroll
    for (int j = 0; j < DS; ++j) Ad[j] = -__expf(A_log[j * DI + d]);

    const float c0 = conv_w[d * 4 + 0], c1 = conv_w[d * 4 + 1];
    const float c2 = conv_w[d * 4 + 2], c3 = conv_w[d * 4 + 3];
    const float cb = conv_b[d];
    const float Dd = Dp[d];

    float s[DS];
#pragma unroll
    for (int j = 0; j < DS; ++j) s[j] = 0.0f;
    float h0 = 0.0f, h1 = 0.0f, h2 = 0.0f;  // conv window (left-pad zeros)

    const float* up = xz + (size_t)b * LQ * 4096 + d;   // u = cols 0..2047
    const float* zp = up + 2048;                        // z = cols 2048..4095
    __bf16* yp = yb + (size_t)b * LQ * DI + d;

#pragma unroll 2
    for (int t = 0; t < LQ; ++t) {
        float ur = up[(size_t)t * 4096];
        float u  = c0 * h0 + c1 * h1 + c2 * h2 + c3 * ur + cb;
        h0 = h1; h1 = h2; h2 = ur;

        float accv = Dd * u;
#pragma unroll
        for (int j = 0; j < DS; ++j) { s[j] = s[j] * Ad[j] + u; accv += s[j]; }

        float z = zp[(size_t)t * 4096];
        float g = z / (1.0f + __expf(-z));       // silu(z)
        yp[(size_t)t * DI] = (__bf16)(accv * g);
    }
}

// ---------------------------------------------------------------------------
extern "C" void kernel_launch(void* const* d_in, const int* in_sizes, int n_in,
                              void* d_out, int out_size, void* d_ws, size_t ws_size,
                              hipStream_t stream) {
    const float* x      = (const float*)d_in[0];  // (4,2048,1024)
    const float* W_in   = (const float*)d_in[1];  // (1024,4096)
    const float* conv_w = (const float*)d_in[2];  // (2048,1,4)
    const float* conv_b = (const float*)d_in[3];  // (2048,)
    const float* A_log  = (const float*)d_in[4];  // (16,2048)
    const float* Dp     = (const float*)d_in[5];  // (2048,)
    const float* W_out  = (const float*)d_in[6];  // (2048,1024)
    float* out = (float*)d_out;                   // (4,2048,1024) f32

    char* ws = (char*)d_ws;
    __bf16* xb    = (__bf16*)(ws);                        // 16 MiB: x as bf16
    __bf16* wtin  = (__bf16*)(ws + (16u  << 20));         //  8 MiB: W_in^T bf16 [4096][1024]
    __bf16* wtout = (__bf16*)(ws + (24u  << 20));         //  4 MiB: W_out^T bf16 [1024][2048]
    float*  xz    = (float*) (ws + (28u  << 20));         // 128 MiB: xz f32 [8192][4096]
    __bf16* yb    = (__bf16*)(ws + (156u << 20));         // 32 MiB: gated y bf16 [8192][2048]

    // prep
    f32_to_bf16_k<<<(8192 * 1024) / 256, 256, 0, stream>>>(x, xb, 8192 * 1024);
    transpose_bf16_k<<<(1024 * 4096) / 256, 256, 0, stream>>>(W_in, wtin, 1024, 4096);
    transpose_bf16_k<<<(2048 * 1024) / 256, 256, 0, stream>>>(W_out, wtout, 2048, 1024);

    // GEMM1: xz = x @ W_in   (M=8192, N=4096, K=1024)
    dim3 g1(4096 / BN, 8192 / BM);
    gemm_bf16_wmma<<<g1, 256, 0, stream>>>(xb, wtin, xz, 8192, 4096, 1024);

    // conv + scan + gate -> yb
    mamba_scan_k<<<dim3(DI / 256, 4), 256, 0, stream>>>(xz, conv_w, conv_b, A_log, Dp, yb);

    // GEMM2: out = y @ W_out (M=8192, N=1024, K=2048)
    dim3 g2(1024 / BN, 8192 / BM);
    gemm_bf16_wmma<<<g2, 256, 0, stream>>>(yb, wtout, out, 8192, 1024, 2048);
}
